// MoRALinear_89266600280130
// MI455X (gfx1250) — matrix operand order repaired and checked by reading, not compiled, not run on hardware
//
#include <hip/hip_runtime.h>
#include <hip/hip_bf16.h>

// MoRA fold: y = x @ (W + tile(A)).T + b  -- one bf16 WMMA GEMM.
// Data staging via CDNA5 Tensor Data Mover (tensor_load_to_lds), double-buffered LDS.
// Workspace layout: [0,128MB) x in bf16; [128MB,160MB) W_eff in bf16. (ws >= 160MB)

typedef __bf16 bf16_t;
typedef __attribute__((ext_vector_type(16))) __bf16 v16bf;
typedef __attribute__((ext_vector_type(8)))  __bf16 v8bf;
typedef __attribute__((ext_vector_type(8)))  float  v8f;
typedef __attribute__((ext_vector_type(4)))  float  v4f;
typedef __attribute__((ext_vector_type(4)))  unsigned int u32x4;
typedef __attribute__((ext_vector_type(8)))  int i32x8;
typedef __attribute__((ext_vector_type(4)))  int i32x4;

#define M_TOT 16384   // B*S
#define N_TOT 4096    // OUT_F
#define K_TOT 4096    // IN_F
#define R_DIM 1024

#define BM 128
#define BN 128
#define BK 64
#define LDSK 72       // BK + 8 halves: TDM pads 4 DWORDs (16B) per 32-DWORD (128B) row

// ---------- pass 1: x fp32 -> bf16 (128MB, L2-resident afterwards) ----------
__global__ __launch_bounds__(256) void cvt_x_bf16(const float* __restrict__ x,
                                                  bf16_t* __restrict__ xb, int n) {
  int idx = (blockIdx.x * 256 + threadIdx.x) * 8;
  if (idx >= n) return;
  v4f a = *(const v4f*)(x + idx);
  v4f c = *(const v4f*)(x + idx + 4);
  v8bf o;
  o[0] = (bf16_t)a[0]; o[1] = (bf16_t)a[1]; o[2] = (bf16_t)a[2]; o[3] = (bf16_t)a[3];
  o[4] = (bf16_t)c[0]; o[5] = (bf16_t)c[1]; o[6] = (bf16_t)c[2]; o[7] = (bf16_t)c[3];
  *(v8bf*)(xb + idx) = o;
}

// ---------- pass 2: W_eff[o,i] = W[o,i] + A[o&1023, i&1023]  (bf16) ----------
__global__ __launch_bounds__(256) void build_weff(const float* __restrict__ W,
                                                  const float* __restrict__ A,
                                                  bf16_t* __restrict__ weff, int n) {
  int idx = (blockIdx.x * 256 + threadIdx.x) * 8;
  if (idx >= n) return;
  int o = idx >> 12;            // / 4096
  int i = idx & (K_TOT - 1);
  const float* arow = A + (size_t)(o & (R_DIM - 1)) * R_DIM + (i & (R_DIM - 1));
  v4f w0 = *(const v4f*)(W + idx);
  v4f w1 = *(const v4f*)(W + idx + 4);
  v4f a0 = *(const v4f*)(arow);
  v4f a1 = *(const v4f*)(arow + 4);
  v8bf ov;
  ov[0] = (bf16_t)(w0[0] + a0[0]); ov[1] = (bf16_t)(w0[1] + a0[1]);
  ov[2] = (bf16_t)(w0[2] + a0[2]); ov[3] = (bf16_t)(w0[3] + a0[3]);
  ov[4] = (bf16_t)(w1[0] + a1[0]); ov[5] = (bf16_t)(w1[1] + a1[1]);
  ov[6] = (bf16_t)(w1[2] + a1[2]); ov[7] = (bf16_t)(w1[3] + a1[3]);
  *(v8bf*)(weff + idx) = ov;
}

// ---------- TDM: load a 128 x BK bf16 tile (row stride K_TOT) into LDS ----------
// D# per ISA 8.3-8.6: 2D tensor, data_size=2B, tile_dim0=BK, tile_dim1=128,
// pad_enable: +4 DWORDs (16B) after every 32 DWORDs (128B = one row) -> LDSK stride.
__device__ __forceinline__ void tdm_load_tile(const bf16_t* gtile, unsigned lds_off,
                                              unsigned rows_total) {
  unsigned long long ga = (unsigned long long)(uintptr_t)gtile;
  u32x4 g0;
  g0[0] = 1u;                                            // count=1 user descriptor
  g0[1] = lds_off;                                       // lds_addr (bytes)
  g0[2] = (unsigned)(ga & 0xffffffffu);                  // global_addr[31:0]
  g0[3] = (unsigned)((ga >> 32) & 0x01ffffffu) | (2u << 30); // global_addr[56:32] | type=2
  i32x8 g1;
  g1[0] = (int)((1u << 16)        // data_size = 2 bytes
              | (1u << 20)        // pad_enable
              | (4u << 22)        // pad_interval: 32 DWORDs (128B row)
              | (3u << 25));      // pad_amount:   4 DWORDs (16B)
  g1[1] = (int)(((unsigned)K_TOT & 0xffffu) << 16);      // tensor_dim0[15:0]
  g1[2] = (int)((rows_total & 0xffffu) << 16);           // dim0[31:16]=0 | tensor_dim1[15:0]
  g1[3] = (int)((rows_total >> 16) | ((unsigned)BK << 16)); // dim1[31:16] | tile_dim0
  g1[4] = BM;                                            // tile_dim1=128 | tile_dim2=0
  g1[5] = K_TOT;                                         // tensor_dim0_stride (elements)
  g1[6] = 0;
  g1[7] = 0;
  i32x4 z4 = {0, 0, 0, 0};
  i32x8 z8 = {0, 0, 0, 0, 0, 0, 0, 0};
  __builtin_amdgcn_tensor_load_to_lds(g0, g1, z4, z4, z8, 0);
}

// ---------- pass 3: C[M,N] = Xb[M,K] @ Weff[N,K]^T + bias ----------
__global__ __launch_bounds__(256)
void gemm_bf16_wmma(const bf16_t* __restrict__ Xb, const bf16_t* __restrict__ Wb,
                    const float* __restrict__ bias, float* __restrict__ out) {
  __shared__ bf16_t As[2][BM][LDSK];
  __shared__ bf16_t Bs[2][BN][LDSK];

  const int tid  = threadIdx.x;
  const int lane = tid & 31;
  const int wid  = tid >> 5;
  const int wm   = wid & 3;    // 4 waves along M
  const int wn   = wid >> 2;   // 2 waves along N

  const int m0 = blockIdx.y * BM;
  const int n0 = blockIdx.x * BN;

  v8f acc[2][4] = {};          // wave tile: 32(M) x 64(N)

  const int half = lane >> 4;  // ISA 7.12.2 half-wave selector
  const int lr   = lane & 15;

  const int NSTEP = K_TOT / BK;   // 64

  if (wid == 0) {   // prologue: prefetch first tiles into buffer 0 (2 TDM ops)
    tdm_load_tile(Xb + (size_t)m0 * K_TOT, (unsigned)(uintptr_t)&As[0][0][0], M_TOT);
    tdm_load_tile(Wb + (size_t)n0 * K_TOT, (unsigned)(uintptr_t)&Bs[0][0][0], N_TOT);
  }

  for (int s = 0; s < NSTEP; ++s) {
    const int kt  = s * BK;
    const int buf = s & 1;

    if (wid == 0) {
      if (s + 1 < NSTEP) {  // prefetch next tiles into other buffer, wait for current
        tdm_load_tile(Xb + (size_t)m0 * K_TOT + (kt + BK),
                      (unsigned)(uintptr_t)&As[buf ^ 1][0][0], M_TOT);
        tdm_load_tile(Wb + (size_t)n0 * K_TOT + (kt + BK),
                      (unsigned)(uintptr_t)&Bs[buf ^ 1][0][0], N_TOT);
        __builtin_amdgcn_s_wait_tensorcnt(2);
      } else {
        __builtin_amdgcn_s_wait_tensorcnt(0);
      }
    }
    __syncthreads();   // current-buffer tiles visible to all waves

#pragma unroll
    for (int kk = 0; kk < BK; kk += 32) {
      // A frag (16x32): lane<16 -> row M=lr, K=kk+{0..7,16..23}; lane>=16 -> +8
      v16bf a_frag[2];
#pragma unroll
      for (int i = 0; i < 2; ++i) {
        const bf16_t* p = &As[buf][wm * 32 + i * 16 + lr][kk + half * 8];
        v8bf lo = *(const v8bf*)p;
        v8bf hi = *(const v8bf*)(p + 16);
        a_frag[i] = __builtin_shufflevector(lo, hi, 0,1,2,3,4,5,6,7,8,9,10,11,12,13,14,15);
      }
      // B frag (32x16): lane<16 -> col N=lr, K=kk+{0..7,16..23}; lane>=16 -> +8
      v16bf b_frag[4];
#pragma unroll
      for (int j = 0; j < 4; ++j) {
        const bf16_t* p = &Bs[buf][wn * 64 + j * 16 + lr][kk + half * 8];
        v8bf lo = *(const v8bf*)p;
        v8bf hi = *(const v8bf*)(p + 16);
        b_frag[j] = __builtin_shufflevector(lo, hi, 0,1,2,3,4,5,6,7,8,9,10,11,12,13,14,15);
      }

#pragma unroll
      for (int i = 0; i < 2; ++i)
#pragma unroll
        for (int j = 0; j < 4; ++j)
          acc[i][j] = __builtin_amdgcn_wmma_f32_16x16x32_bf16(
              /*neg_a=*/false, a_frag[i], /*neg_b=*/false, b_frag[j],
              /*c_mod=*/(short)0, acc[i][j], /*reuse_a=*/false, /*reuse_b=*/false);
    }
    __syncthreads();   // all reads of current buffer done before TDM overwrites it
  }

  // Epilogue: D vgpr r holds M = r + half*8, N = lr  (ISA 7.12.2)
#pragma unroll
  for (int j = 0; j < 4; ++j) {
    int col  = n0 + wn * 64 + j * 16 + lr;
    float bv = bias[col];
#pragma unroll
    for (int i = 0; i < 2; ++i) {
      int mrow  = m0 + wm * 32 + i * 16 + half * 8;
      float* po = out + (size_t)mrow * N_TOT + col;
#pragma unroll
      for (int r = 0; r < 8; ++r)
        po[(size_t)r * N_TOT] = acc[i][j][r] + bv;
    }
  }
}

extern "C" void kernel_launch(void* const* d_in, const int* in_sizes, int n_in,
                              void* d_out, int out_size, void* d_ws, size_t ws_size,
                              hipStream_t stream) {
  const float* x = (const float*)d_in[0];   // [4,4096,4096]
  const float* W = (const float*)d_in[1];   // [4096,4096]
  const float* b = (const float*)d_in[2];   // [4096]
  const float* A = (const float*)d_in[3];   // [1024,1024]
  float* out = (float*)d_out;               // [4,4096,4096]

  bf16_t* xb   = (bf16_t*)d_ws;                                      // 128MB
  bf16_t* weff = (bf16_t*)((char*)d_ws + (size_t)M_TOT * K_TOT * 2); // +32MB

  const int nx = M_TOT * K_TOT;   // 64M elems
  cvt_x_bf16<<<nx / (256 * 8), 256, 0, stream>>>(x, xb, nx);

  const int nw = N_TOT * K_TOT;   // 16M elems
  build_weff<<<nw / (256 * 8), 256, 0, stream>>>(W, A, weff, nw);

  dim3 grid(N_TOT / BN, M_TOT / BM);  // 32 x 128 blocks
  gemm_bf16_wmma<<<grid, 256, 0, stream>>>(xb, weff, b, out);
}
